// GATLayer_62311385531064
// MI455X (gfx1250) — matrix operand order repaired and checked by reading, not compiled, not run on hardware
//
#include <hip/hip_runtime.h>

typedef __attribute__((ext_vector_type(16))) _Float16 v16h;
typedef __attribute__((ext_vector_type(8)))  float    v8f;

#define NN    8192
#define INF_  128
#define OUTF  64
#define ALPHA 0.2f
#define LOG2E 1.4426950408889634f
#define NWAVES 8

// ---------------------------------------------------------------------------
// Kernel 1: per 32-row tile compute Wh = in @ W, s_src/s_dst (pre-scaled by
// log2(e)), and repack Wh into f16 WMMA-B lane layout.
// ---------------------------------------------------------------------------
__global__ __launch_bounds__(256) void k_prep(const float* __restrict__ in,
                                              const float* __restrict__ W,
                                              const float* __restrict__ a,
                                              float* __restrict__ ssrc,
                                              float* __restrict__ sdst,
                                              v16h*  __restrict__ whb)
{
    __shared__ float intile[32][INF_];     // 16 KB
    __shared__ float wh[32][65];           // padded to kill bank conflicts

    const int t  = threadIdx.x;
    const int b  = blockIdx.x;
    const int r0 = b * 32;

    // stage 32x128 input tile (coalesced float4)
    const float4* inv4 = (const float4*)(in + (size_t)r0 * INF_);
    float4* itv4 = (float4*)(&intile[0][0]);
    for (int i = t; i < 32 * INF_ / 4; i += 256) itv4[i] = inv4[i];
    __syncthreads();

    // Wh tile: thread -> col j = t&63, rows tr, tr+4, ..., tr+28
    const int j  = t & 63;
    const int tr = t >> 6;
    float acc[8];
#pragma unroll
    for (int rr = 0; rr < 8; ++rr) acc[rr] = 0.f;
    for (int k = 0; k < INF_; ++k) {
        float wk = W[k * OUTF + j];
#pragma unroll
        for (int rr = 0; rr < 8; ++rr) acc[rr] += intile[rr * 4 + tr][k] * wk;
    }
#pragma unroll
    for (int rr = 0; rr < 8; ++rr) wh[rr * 4 + tr][j] = acc[rr];
    __syncthreads();

    // attention logits (pre-scaled by log2(e); lrelu commutes with +scale)
    if (t < 32) {
        float s1 = 0.f, s2 = 0.f;
        for (int jj = 0; jj < OUTF; ++jj) {
            float v = wh[t][jj];
            s1 += v * a[jj];
            s2 += v * a[OUTF + jj];
        }
        ssrc[r0 + t] = s1 * LOG2E;
        sdst[r0 + t] = s2 * LOG2E;
    }

    // repack 32x64 Wh tile -> 4 WMMA-B tiles (32x16 f16), lane layout:
    // N=lane%16, h=lane/16: K=8h+e (e<8), K=16+8h+(e-8) (e>=8)
    if (t < 128) {
        const int t4   = t >> 5;
        const int lane = t & 31;
        const int n    = lane & 15;
        const int koff = (lane & 16) >> 1; // 0 or 8
        v16h hv;
#pragma unroll
        for (int e = 0; e < 16; ++e)
            hv[e] = (_Float16)wh[koff + (e & 7) + ((e & 8) << 1)][t4 * 16 + n];
        whb[((size_t)b * 4 + t4) * 32 + lane] = hv;
    }
}

// ---------------------------------------------------------------------------
// Kernel 1c: S_max = max(s_dst)  (single block reduction, deterministic)
// ---------------------------------------------------------------------------
__global__ __launch_bounds__(256) void k_max(const float* __restrict__ sdst,
                                             float* __restrict__ smax)
{
    __shared__ float red[256];
    const int t = threadIdx.x;
    float m = -3.0e38f;
    for (int i = t; i < NN; i += 256) m = fmaxf(m, sdst[i]);
    red[t] = m;
    __syncthreads();
    for (int s = 128; s > 0; s >>= 1) {
        if (t < s) red[t] = fmaxf(red[t], red[t + s]);
        __syncthreads();
    }
    if (t == 0) smax[0] = red[0];
}

// ---------------------------------------------------------------------------
// Kernel 2: fused masked-softmax + att@Wh, single pass over adj (256 MB).
// 512 blocks x 256 threads (8 waves): each wave owns the block's 16 rows and
// every 8th 32-column chunk -> 4096 waves to hide L2/HBM latency.
// Row sums come from a 5th WMMA against an all-ones B (co-executes w/ VALU).
// ---------------------------------------------------------------------------
__device__ __forceinline__ float pfun(float av, float sd, float ss, float mrow)
{
    float x = ss + sd;                       // scaled logit
    x = fmaxf(x, ALPHA * x);                 // leaky_relu (alpha<1)
    float e = __builtin_amdgcn_exp2f(x - mrow);  // <= 1, no overflow
    return av > 0.f ? e : 0.f;               // adjacency mask
}

__global__ __launch_bounds__(32 * NWAVES) void k_gat(const float* __restrict__ adj,
                                                     const float* __restrict__ ssrc,
                                                     const float* __restrict__ sdst,
                                                     const v16h*  __restrict__ whb,
                                                     const float* __restrict__ smaxp,
                                                     float* __restrict__ out)
{
    __shared__ float red[NWAVES][16][64];   // 32 KB partial O per wave
    __shared__ float lrow[NWAVES][16];      // per-wave row-sum partials
    __shared__ float rinv[16];

    const int lane = threadIdx.x & 31;
    // wave id as a scalar -> chunk counter / B base stay in SGPRs
    const int w    = __builtin_amdgcn_readfirstlane(threadIdx.x) >> 5;
    const int row0 = blockIdx.x * 16;
    const int row  = row0 + (lane & 15);        // A-matrix row M
    const int koff = (lane & 16) >> 1;          // 0 or 8: lane-half K offset

    const float smax = smaxp[0];
    const float ss   = ssrc[row];
    float mb = ss + smax;
    const float mrow = fmaxf(mb, ALPHA * mb);   // per-row softmax shift (upper bound)

    const float* __restrict__ arow  = adj + (size_t)row * NN;
    const v16h*  __restrict__ bbase = whb + lane;

    // all-ones B matrix: row-sum WMMA operand (constant, no memory traffic)
    v16h ones;
#pragma unroll
    for (int e = 0; e < 16; ++e) ones[e] = (_Float16)1.0f;

    v8f acc0 = {}, acc1 = {}, acc2 = {}, acc3 = {}, accL = {};

    for (int c = w; c < NN / 32; c += NWAVES) {
        const int cb = c << 5;

        // ---- all long-latency loads issued up front ----
        // adj tile: lanes<16 cover K {0..7,16..23}, lanes>=16 {8..15,24..31};
        // two lanes cover exactly one 128B line per row -> fully coalesced.
        float4 av0 = *(const float4*)(arow + cb + koff);
        float4 av1 = *(const float4*)(arow + cb + koff + 4);
        float4 av2 = *(const float4*)(arow + cb + koff + 16);
        float4 av3 = *(const float4*)(arow + cb + koff + 20);
        float4 sv0 = *(const float4*)(sdst + cb + koff);
        float4 sv1 = *(const float4*)(sdst + cb + koff + 4);
        float4 sv2 = *(const float4*)(sdst + cb + koff + 16);
        float4 sv3 = *(const float4*)(sdst + cb + koff + 20);

        // pre-packed B tiles: 2x global_load_b128 each, L2-resident
        const v16h* bp = bbase + (size_t)c * 128;
        v16h b0 = bp[0];
        v16h b1 = bp[32];
        v16h b2 = bp[64];
        v16h b3 = bp[96];

        // speculative prefetch of this wave's next adj chunk into GL2
        __builtin_prefetch(arow + cb + 32 * NWAVES + koff, 0, 3);

        // ---- elementwise mask + exp (overlaps load latency) ----
        v16h A;
        A[0]  = (_Float16)pfun(av0.x, sv0.x, ss, mrow);
        A[1]  = (_Float16)pfun(av0.y, sv0.y, ss, mrow);
        A[2]  = (_Float16)pfun(av0.z, sv0.z, ss, mrow);
        A[3]  = (_Float16)pfun(av0.w, sv0.w, ss, mrow);
        A[4]  = (_Float16)pfun(av1.x, sv1.x, ss, mrow);
        A[5]  = (_Float16)pfun(av1.y, sv1.y, ss, mrow);
        A[6]  = (_Float16)pfun(av1.z, sv1.z, ss, mrow);
        A[7]  = (_Float16)pfun(av1.w, sv1.w, ss, mrow);
        A[8]  = (_Float16)pfun(av2.x, sv2.x, ss, mrow);
        A[9]  = (_Float16)pfun(av2.y, sv2.y, ss, mrow);
        A[10] = (_Float16)pfun(av2.z, sv2.z, ss, mrow);
        A[11] = (_Float16)pfun(av2.w, sv2.w, ss, mrow);
        A[12] = (_Float16)pfun(av3.x, sv3.x, ss, mrow);
        A[13] = (_Float16)pfun(av3.y, sv3.y, ss, mrow);
        A[14] = (_Float16)pfun(av3.z, sv3.z, ss, mrow);
        A[15] = (_Float16)pfun(av3.w, sv3.w, ss, mrow);

        // ---- matrix ops: 4 output N-tiles + row-sum tile ----
        acc0 = __builtin_amdgcn_wmma_f32_16x16x32_f16(false, A, false, b0,   (short)0, acc0, false, false);
        acc1 = __builtin_amdgcn_wmma_f32_16x16x32_f16(false, A, false, b1,   (short)0, acc1, false, false);
        acc2 = __builtin_amdgcn_wmma_f32_16x16x32_f16(false, A, false, b2,   (short)0, acc2, false, false);
        acc3 = __builtin_amdgcn_wmma_f32_16x16x32_f16(false, A, false, b3,   (short)0, acc3, false, false);
        accL = __builtin_amdgcn_wmma_f32_16x16x32_f16(false, A, false, ones, (short)0, accL, false, false);
    }

    // spill per-wave partial O (C/D layout: VGPR r, lane l -> M=r+8*(l>=16), N=l%16)
    const int Mhi = (lane & 16) >> 1;
    const int n   = lane & 15;
#pragma unroll
    for (int r = 0; r < 8; ++r) {
        red[w][r + Mhi][n]      = acc0[r];
        red[w][r + Mhi][16 + n] = acc1[r];
        red[w][r + Mhi][32 + n] = acc2[r];
        red[w][r + Mhi][48 + n] = acc3[r];
    }
    // accL: every column holds the same row sum; publish from n==0 lanes
    if (n == 0) {
#pragma unroll
        for (int r = 0; r < 8; ++r) lrow[w][r + Mhi] = accL[r];
    }
    __syncthreads();

    if (threadIdx.x < 16) {
        float s = 0.f;
#pragma unroll
        for (int ww = 0; ww < NWAVES; ++ww) s += lrow[ww][threadIdx.x];
        rinv[threadIdx.x] = 1.0f / s;   // self-loop guarantees s > 0
    }
    __syncthreads();

    for (int idx = threadIdx.x; idx < 16 * 64; idx += 32 * NWAVES) {
        int M = idx >> 6, f = idx & 63;
        float o = 0.f;
#pragma unroll
        for (int ww = 0; ww < NWAVES; ++ww) o += red[ww][M][f];
        out[(size_t)(row0 + M) * OUTF + f] = o * rinv[M];
    }
}

// ---------------------------------------------------------------------------
// Host launcher
// ---------------------------------------------------------------------------
extern "C" void kernel_launch(void* const* d_in, const int* in_sizes, int n_in,
                              void* d_out, int out_size, void* d_ws, size_t ws_size,
                              hipStream_t stream)
{
    const float* input_h = (const float*)d_in[0];   // 8192x128
    const float* adj     = (const float*)d_in[1];   // 8192x8192
    const float* W       = (const float*)d_in[2];   // 128x64
    const float* a       = (const float*)d_in[3];   // 128x1
    float* out = (float*)d_out;                     // 8192x64

    char* ws = (char*)d_ws;
    float* smax = (float*)ws;                         // 4 B (padded to 256)
    float* ssrc = (float*)(ws + 256);                 // 32 KB
    float* sdst = (float*)(ws + 256 + 32768);         // 32 KB
    v16h*  whb  = (v16h*)(ws + 256 + 65536);          // 1 MB packed f16 B tiles

    k_prep<<<NN / 32, 256, 0, stream>>>(input_h, W, a, ssrc, sdst, whb);
    k_max<<<1, 256, 0, stream>>>(sdst, smax);
    k_gat<<<NN / 16, 32 * NWAVES, 0, stream>>>(adj, ssrc, sdst, whb, smax, out);
}
